// FiLMSAtten2d_7756710936625
// MI455X (gfx1250) — matrix operand order repaired and checked by reading, not compiled, not run on hardware
//
#include <hip/hip_runtime.h>
#include <hip/hip_bf16.h>

typedef _Float16 half_t;
typedef __attribute__((ext_vector_type(16))) _Float16 v16h;
typedef __attribute__((ext_vector_type(8)))  _Float16 v8h;
typedef __attribute__((ext_vector_type(8)))  float    v8f;

// Async global->LDS DMA path (gfx1250 only; host pass never codegens it)
#if defined(__gfx1250__)
#  define USE_ASYNC_LDS 1
#endif

// ---- problem constants ----
constexpr int BB   = 2;
constexpr int CC   = 256;
constexpr int HH   = 64;
constexpr int WW   = 64;
constexpr int NN   = 4096;      // H*W
constexpr int FEAT = 256;
constexpr int HEADS= 8;
constexpr int HS   = 32;
constexpr int OUTC = 256;
constexpr int QKVC = 3 * FEAT;  // 768
constexpr float QSCALE = 0.17677669529663688f; // 1/sqrt(32)

// ---- workspace layout (bytes) ----
constexpr size_t XP_OFF = 0;
constexpr size_t XP_SZ  = (size_t)BB * NN * CC * sizeof(half_t);        // 4 MiB  [b][n][c]
constexpr size_t WQ_OFF = XP_OFF + XP_SZ;
constexpr size_t WQ_SZ  = (size_t)QKVC * CC * sizeof(half_t);
constexpr size_t WP_OFF = WQ_OFF + WQ_SZ;
constexpr size_t WP_SZ  = (size_t)OUTC * FEAT * sizeof(half_t);
constexpr size_t Q_OFF  = WP_OFF + WP_SZ;
constexpr size_t QKV1_SZ= (size_t)BB * HEADS * NN * HS * sizeof(half_t); // 4 MiB
constexpr size_t K_OFF  = Q_OFF + QKV1_SZ;
constexpr size_t V_OFF  = K_OFF + QKV1_SZ;
constexpr size_t AT_OFF = V_OFF + QKV1_SZ;

// ============ WMMA helpers (v_wmma_f32_16x16x32_f16) ============
__device__ __forceinline__ v8f wmma16(v16h a, v16h b, v8f c) {
    return __builtin_amdgcn_wmma_f32_16x16x32_f16(
        /*neg_a=*/false, a, /*neg_b=*/false, b,
        /*c_mod=*/(short)0, c, /*reuse_a=*/false, /*reuse_b=*/false);
}

// A fragment 16x32 (M x K) from row-major [m][k] storage, ld in halfs.
__device__ __forceinline__ v16h load_frag_a(const half_t* p, int ld) {
    int lane = threadIdx.x & 31;
    int row  = lane & 15;
    int kb   = (lane >> 4) << 3;           // 0 or 8
    const half_t* r = p + (size_t)row * ld;
    v8h lo = *(const v8h*)(r + kb);
    v8h hi = *(const v8h*)(r + 16 + kb);
    v16h a;
#pragma unroll
    for (int i = 0; i < 8; ++i) { a[i] = lo[i]; a[8 + i] = hi[i]; }
    return a;
}

// B fragment 32x16 (K x N) from storage laid out as [N][K] rows (ld in halfs).
__device__ __forceinline__ v16h load_frag_b_nt(const half_t* p, int ld) {
    int lane = threadIdx.x & 31;
    const half_t* r = p + (size_t)(lane & 15) * ld + ((lane >> 4) << 3);
    v8h lo = *(const v8h*)(r);
    v8h hi = *(const v8h*)(r + 16);
    v16h b;
#pragma unroll
    for (int i = 0; i < 8; ++i) { b[i] = lo[i]; b[8 + i] = hi[i]; }
    return b;
}

// 32-byte global -> LDS copy: async DMA (ASYNCcnt) on gfx1250.
// INST_OFFSET is added to both the LDS and global addresses (ISA 15.18.3),
// so one lds-offset VGPR + one 64-bit address pair covers both beats.
__device__ __forceinline__ void stage32(half_t* dst_lds, const half_t* src_global) {
#ifdef USE_ASYNC_LDS
    uint32_t loff = (uint32_t)(uintptr_t)dst_lds;   // LDS byte offset = addr[31:0]
    asm volatile("global_load_async_to_lds_b128 %0, %1, off\n\t"
                 "global_load_async_to_lds_b128 %0, %1, off offset:16"
                 :: "v"(loff), "v"(src_global)
                 : "memory");
#else
    v8h t0 = ((const v8h*)src_global)[0];
    v8h t1 = ((const v8h*)src_global)[1];
    ((v8h*)dst_lds)[0] = t0;
    ((v8h*)dst_lds)[1] = t1;
#endif
}
__device__ __forceinline__ void stage_wait() {
#ifdef USE_ASYNC_LDS
    asm volatile("s_wait_asynccnt 0" ::: "memory");
#else
    asm volatile("s_wait_dscnt 0" ::: "memory");
#endif
}

// 16-lane-group reductions (rows of a WMMA C fragment live in half-wave groups)
__device__ __forceinline__ float rowmax16(float v) {
    v = fmaxf(v, __shfl_xor(v, 1, 16));
    v = fmaxf(v, __shfl_xor(v, 2, 16));
    v = fmaxf(v, __shfl_xor(v, 4, 16));
    v = fmaxf(v, __shfl_xor(v, 8, 16));
    return v;
}
__device__ __forceinline__ float rowsum16(float v) {
    v += __shfl_xor(v, 1, 16);
    v += __shfl_xor(v, 2, 16);
    v += __shfl_xor(v, 4, 16);
    v += __shfl_xor(v, 8, 16);
    return v;
}

// ============ kernel 0a: f32 -> f16 cast ============
__global__ void cvt_f16_kernel(const float* __restrict__ src, half_t* __restrict__ dst, int n) {
    int i = blockIdx.x * blockDim.x + threadIdx.x;
    if (i < n) dst[i] = (half_t)src[i];
}

// ============ kernel 0b: fused SiLU pre-activation, transposed to [b][n][c] f16 ============
__global__ __launch_bounds__(256) void preact_tr_kernel(
    const float* __restrict__ x,              // [B][C][N]
    const float* __restrict__ pre_act_bias,   // [C]
    const float* __restrict__ pre_qkv_bias,   // [C]
    half_t* __restrict__ xp)                  // [B][N][C]
{
    __shared__ float tile[64][65];
    int nb  = blockIdx.x % (NN / 64);
    int tmp = blockIdx.x / (NN / 64);
    int cb  = tmp % (CC / 64);
    int b   = tmp / (CC / 64);
    int c0 = cb * 64, n0 = nb * 64;

    int t = threadIdx.x;
    {   // read 64 rows of x (fixed c, 64 n), fused silu+bias
        int r   = t >> 2;            // c row 0..63
        int off = (t & 3) * 16;      // n offset
        int c   = c0 + r;
        float pab = pre_act_bias[c], pqb = pre_qkv_bias[c];
        const float* src = x + ((size_t)b * CC + c) * NN + n0 + off;
#pragma unroll
        for (int i = 0; i < 16; ++i) {
            float v = src[i] + pab;
            tile[r][off + i] = v / (1.0f + __expf(-v)) + pqb;
        }
    }
    __syncthreads();
    {   // write transposed: fixed n row, 16 consecutive c
        int nrow = t >> 2;
        int cseg = (t & 3) * 16;
        half_t* dst = xp + ((size_t)b * NN + n0 + nrow) * CC + c0 + cseg;
        v8h o0, o1;
#pragma unroll
        for (int i = 0; i < 8; ++i) {
            o0[i] = (half_t)tile[cseg + i][nrow];
            o1[i] = (half_t)tile[cseg + 8 + i][nrow];
        }
        ((v8h*)dst)[0] = o0;
        ((v8h*)dst)[1] = o1;
    }
}

// ============ kernel 1: QKV GEMM (768 x 4096 x 256 per batch) + epilogue ============
// 128 threads (4 waves); tile 64(M) x 64(N); A direct from global; B tile
// double-buffered in LDS via async global->LDS DMA overlapped with WMMA.
__global__ __launch_bounds__(128) void qkv_gemm_kernel(
    const half_t* __restrict__ wq,      // [768][256] f16
    const half_t* __restrict__ xp,      // [B][N][C] f16 (transposed preact)
    const float*  __restrict__ b_qkv,   // [768]
    const float*  __restrict__ peqh, const float* __restrict__ peqw,
    const float*  __restrict__ pekh, const float* __restrict__ pekw,
    const float*  __restrict__ modm, const float* __restrict__ modb,
    half_t* __restrict__ Qo,            // [B][HEADS][N][HS]
    half_t* __restrict__ Ko,            // [B][HEADS][N][HS]
    half_t* __restrict__ Vo)            // [B][HEADS][HS][N]
{
    __shared__ half_t lB[2][64 * 32];   // double-buffered [n][k] tiles

    int ntiles = NN / 64;
    int nb  = blockIdx.x % ntiles;
    int tmp = blockIdx.x / ntiles;
    int mb  = tmp % (QKVC / 64);
    int b   = tmp / (QKVC / 64);
    int o0  = mb * 64, n0 = nb * 64;

    int t    = threadIdx.x;
    int lane = t & 31, wv = t >> 5;

    // per-thread stage source row (16 contiguous halfs) and LDS slot
    const half_t* xrow  = xp + ((size_t)b * NN + n0 + (t >> 1)) * CC + (t & 1) * 16;
    const int     lslot = (t >> 1) * 32 + (t & 1) * 16;

    v8f zero = {};
    v8f acc[4] = {zero, zero, zero, zero};

    // prologue: stage k-slice 0 into buffer 0
    stage32(&lB[0][lslot], xrow);
    stage_wait();
    __syncthreads();

    for (int kk = 0; kk < CC; kk += 32) {
        int buf = (kk >> 5) & 1;
        if (kk + 32 < CC)                       // overlap DMA of next slice with WMMAs
            stage32(&lB[buf ^ 1][lslot], xrow + kk + 32);

        v16h af = load_frag_a(wq + (size_t)(o0 + wv * 16) * CC + kk, CC);  // direct global
#pragma unroll
        for (int nt = 0; nt < 4; ++nt) {
            v16h bf = load_frag_b_nt(&lB[buf][nt * 16 * 32], 32);
            acc[nt] = wmma16(af, bf, acc[nt]);
        }
        stage_wait();
        __syncthreads();
    }

    // epilogue: bias + PE + FiLM, scatter into Q/K/V f16 buffers
    int colb = lane & 15, rb = (lane >> 4) << 3;
#pragma unroll
    for (int nt = 0; nt < 4; ++nt) {
#pragma unroll
        for (int j = 0; j < 8; ++j) {
            int o = o0 + wv * 16 + rb + j;
            int n = n0 + nt * 16 + colb;
            float val = acc[nt][j] + b_qkv[o];
            int hrow = n >> 6, wcol = n & 63;
            if (o < FEAT) {                             // Q
                int f = o;
                val += peqh[f * HH + hrow] + peqw[f * WW + wcol];
                val  = val * modm[b * FEAT + f] + modb[b * FEAT + f];
                val *= QSCALE;                          // fold 1/sqrt(HS)
                Qo[(((size_t)b * HEADS + (f >> 5)) * NN + n) * HS + (f & 31)] = (half_t)val;
            } else if (o < 2 * FEAT) {                  // K
                int f = o - FEAT;
                val += pekh[f * HH + hrow] + pekw[f * WW + wcol];
                Ko[(((size_t)b * HEADS + (f >> 5)) * NN + n) * HS + (f & 31)] = (half_t)val;
            } else {                                    // V (transposed: [d][n])
                int f = o - 2 * FEAT;
                Vo[(((size_t)b * HEADS + (f >> 5)) * HS + (f & 31)) * NN + n] = (half_t)val;
            }
        }
    }
}

// ============ kernel 2: flash attention per (b,head) ============
// 128 threads (4 waves); each wave owns 16 query rows; stream keys 64 at a time:
// 4 S-WMMAs -> online softmax -> P via wave-private LDS -> 4 PV-WMMAs.
__global__ __launch_bounds__(128) void attn_kernel(
    const half_t* __restrict__ Qi,      // [BH][N][HS]
    const half_t* __restrict__ Ki,      // [BH][N][HS]
    const half_t* __restrict__ Vi,      // [BH][HS][N]
    const float*  __restrict__ ppb,     // [FEAT] pre_proj_bias
    half_t* __restrict__ ATT)           // [B][N][FEAT]
{
    __shared__ half_t lP[4][16 * 64];   // wave-private P tiles (16 rows x 64 keys)

    int qt = blockIdx.x % (NN / 64);
    int bh = blockIdx.x / (NN / 64);
    int b    = bh / HEADS;
    int head = bh % HEADS;

    int lane = threadIdx.x & 31, wv = threadIdx.x >> 5;
    int n0   = qt * 64 + wv * 16;

    const half_t* Qb = Qi + (size_t)bh * NN * HS;
    const half_t* Kb = Ki + (size_t)bh * NN * HS;
    const half_t* Vb = Vi + (size_t)bh * HS * NN;

    v16h qf = load_frag_a(Qb + (size_t)n0 * HS, HS);

    v8f o0 = {}, o1 = {};
    float rmax[8], rsum[8];
#pragma unroll
    for (int j = 0; j < 8; ++j) { rmax[j] = -1e30f; rsum[j] = 0.0f; }

    half_t* myP = lP[wv];

    for (int m0 = 0; m0 < NN; m0 += 64) {
        if (m0 + 64 < NN) {
            __builtin_prefetch(Kb + (size_t)(m0 + 64) * HS, 0, 1);
            __builtin_prefetch(Vb + m0 + 64, 0, 1);
        }
        // S tiles: logits[n, m0..m0+63]
        v8f z = {};
        v8f s[4];
#pragma unroll
        for (int i = 0; i < 4; ++i)
            s[i] = wmma16(qf, load_frag_b_nt(Kb + (size_t)(m0 + 16 * i) * HS, HS), z);

        // online softmax per row (row j lives in a 16-lane group)
#pragma unroll
        for (int j = 0; j < 8; ++j) {
            float mj = rowmax16(fmaxf(fmaxf(s[0][j], s[1][j]), fmaxf(s[2][j], s[3][j])));
            float nm = fmaxf(rmax[j], mj);
            float sc = __expf(rmax[j] - nm);
            float p0 = __expf(s[0][j] - nm);
            float p1 = __expf(s[1][j] - nm);
            float p2 = __expf(s[2][j] - nm);
            float p3 = __expf(s[3][j] - nm);
            float sj = rowsum16((p0 + p1) + (p2 + p3));
            rsum[j] = rsum[j] * sc + sj;
            o0[j] *= sc;
            o1[j] *= sc;
            rmax[j] = nm;
            int r = j + ((lane >> 4) << 3);
            half_t* pr = myP + r * 64 + (lane & 15);
            pr[0]  = (half_t)p0;
            pr[16] = (half_t)p1;
            pr[32] = (half_t)p2;
            pr[48] = (half_t)p3;
        }
        asm volatile("s_wait_dscnt 0" ::: "memory");

        // O += P @ V : two 32-key subchunks x two d-halves
#pragma unroll
        for (int mc = 0; mc < 2; ++mc) {
            v16h pf = load_frag_a(myP + mc * 32, 64);
            v16h v0 = load_frag_b_nt(Vb + m0 + mc * 32, NN);                   // d = 0..15
            v16h v1 = load_frag_b_nt(Vb + (size_t)16 * NN + m0 + mc * 32, NN); // d = 16..31
            o0 = wmma16(pf, v0, o0);
            o1 = wmma16(pf, v1, o1);
        }
    }

    // normalize + pre_proj_bias, store f16 attn [b][n][f]
    int colb = lane & 15, rb = (lane >> 4) << 3;
#pragma unroll
    for (int j = 0; j < 8; ++j) {
        int n = n0 + rb + j;
        float inv = 1.0f / rsum[j];
        int f0 = head * HS + colb;
        int f1 = f0 + 16;
        ATT[((size_t)b * NN + n) * FEAT + f0] = (half_t)(o0[j] * inv + ppb[f0]);
        ATT[((size_t)b * NN + n) * FEAT + f1] = (half_t)(o1[j] * inv + ppb[f1]);
    }
}

// ============ kernel 3: output projection + residual ============
__global__ __launch_bounds__(128) void proj_kernel(
    const half_t* __restrict__ wp,      // [256][256] f16
    const half_t* __restrict__ ATT,     // [B][N][FEAT] f16
    const float*  __restrict__ x,       // [B][OUT][N]
    const float*  __restrict__ id_scale,// [OUT]
    float* __restrict__ out)            // [B][OUT][N]
{
    int nb  = blockIdx.x % (NN / 64);
    int tmp = blockIdx.x / (NN / 64);
    int mb  = tmp % (OUTC / 64);
    int b   = tmp / (OUTC / 64);

    int lane = threadIdx.x & 31, wv = threadIdx.x >> 5;
    int o0 = mb * 64 + wv * 16;
    int n0 = nb * 64;

    v8f zero = {};
    v8f acc[4] = {zero, zero, zero, zero};

    for (int kk = 0; kk < FEAT; kk += 32) {
        v16h af = load_frag_a(wp + (size_t)o0 * FEAT + kk, FEAT);
#pragma unroll
        for (int nt = 0; nt < 4; ++nt) {
            v16h bf = load_frag_b_nt(ATT + ((size_t)b * NN + n0 + nt * 16) * FEAT + kk, FEAT);
            acc[nt] = wmma16(af, bf, acc[nt]);
        }
    }

    int colb = lane & 15, rb = (lane >> 4) << 3;
#pragma unroll
    for (int nt = 0; nt < 4; ++nt) {
#pragma unroll
        for (int j = 0; j < 8; ++j) {
            int o = o0 + rb + j;
            int n = n0 + nt * 16 + colb;
            size_t idx = ((size_t)b * OUTC + o) * NN + n;
            out[idx] = x[idx] * id_scale[o] + acc[nt][j];
        }
    }
}

// ============ host launcher ============
extern "C" void kernel_launch(void* const* d_in, const int* in_sizes, int n_in,
                              void* d_out, int out_size, void* d_ws, size_t ws_size,
                              hipStream_t stream) {
    const float* x    = (const float*)d_in[0];
    const float* modm = (const float*)d_in[1];
    const float* modb = (const float*)d_in[2];
    const float* wqkv = (const float*)d_in[3];
    const float* bqkv = (const float*)d_in[4];
    const float* wproj= (const float*)d_in[5];
    const float* peqh = (const float*)d_in[6];
    const float* peqw = (const float*)d_in[7];
    const float* pekh = (const float*)d_in[8];
    const float* pekw = (const float*)d_in[9];
    const float* pab  = (const float*)d_in[10];
    const float* pqb  = (const float*)d_in[11];
    const float* ppb  = (const float*)d_in[12];
    const float* idsc = (const float*)d_in[13];
    float* out = (float*)d_out;

    char* ws = (char*)d_ws;
    half_t* XP = (half_t*)(ws + XP_OFF);
    half_t* WQ = (half_t*)(ws + WQ_OFF);
    half_t* WP = (half_t*)(ws + WP_OFF);
    half_t* Qb = (half_t*)(ws + Q_OFF);
    half_t* Kb = (half_t*)(ws + K_OFF);
    half_t* Vb = (half_t*)(ws + V_OFF);
    half_t* AT = (half_t*)(ws + AT_OFF);

    // 0) precision casts + fused transposed preact
    cvt_f16_kernel<<<(QKVC * CC + 255) / 256, 256, 0, stream>>>(wqkv, WQ, QKVC * CC);
    cvt_f16_kernel<<<(OUTC * FEAT + 255) / 256, 256, 0, stream>>>(wproj, WP, OUTC * FEAT);
    preact_tr_kernel<<<BB * (CC / 64) * (NN / 64), 256, 0, stream>>>(x, pab, pqb, XP);
    // 1) QKV projection + PE + FiLM
    qkv_gemm_kernel<<<BB * (QKVC / 64) * (NN / 64), 128, 0, stream>>>(
        WQ, XP, bqkv, peqh, peqw, pekh, pekw, modm, modb, Qb, Kb, Vb);
    // 2) flash attention
    attn_kernel<<<BB * HEADS * (NN / 64), 128, 0, stream>>>(Qb, Kb, Vb, ppb, AT);
    // 3) projection + residual
    proj_kernel<<<BB * (OUTC / 64) * (NN / 64), 128, 0, stream>>>(WP, AT, x, idsc, out);
}
